// Step3Model_34840774705371
// MI455X (gfx1250) — matrix operand order repaired and checked by reading, not compiled, not run on hardware
//
#include <hip/hip_runtime.h>
#include <hip/hip_bf16.h>

typedef __attribute__((ext_vector_type(16))) __bf16 v16bf;
typedef __attribute__((ext_vector_type(8)))  float  v8f;

#define V_    64
#define T_    8192
#define DA_   256
#define DS_   256
#define KRDU_ 512
#define KA_   40
#define EPS_  1e-8f

// output layout (floats, concatenated in reference return order)
#define O_PDET   0
#define O_TAU    1
#define O_YHAT   2
#define O_E      42
#define O_EVIEW  (O_E + V_*T_)        // 524330
#define O_RVIEW  (O_EVIEW + V_)       // 524394
#define O_IVIEW  (O_RVIEW + V_)       // 524458
#define O_STARTS (O_IVIEW + V_)       // 524522
#define O_ENDS   (O_STARTS + V_*T_)   // 1048810
#define O_FLAG   (O_ENDS + V_*T_)     // 1573098

// ---------------------------------------------------------------------------
// Kernel 1: all the small heads. 1 block, 256 threads.
// ws[0..255] = l2-normalized zs ; ws[256] = p_det
// ---------------------------------------------------------------------------
__global__ __launch_bounds__(256) void head_kernel(
    const float* __restrict__ alpha, const float* __restrict__ pi,
    const float* __restrict__ US,    const float* __restrict__ Ws_w,
    const float* __restrict__ Ws_b,  const float* __restrict__ det_w,
    const float* __restrict__ det_b, const float* __restrict__ cls_w,
    const float* __restrict__ cls_b, const float* __restrict__ tau_c0,
    const float* __restrict__ lamU,  const float* __restrict__ lamC,
    const float* __restrict__ lamP,  const float* __restrict__ lamPi,
    const float* __restrict__ nu,    const float* __restrict__ risk,
    float* __restrict__ out, float* __restrict__ ws)
{
    __shared__ float zs[DS_];
    __shared__ float red[256];
    const int tid = threadIdx.x;

    // zs = US @ Ws_w.T + Ws_b  (one output per thread)
    float z = 0.f;
    for (int k = 0; k < KRDU_; ++k) z = fmaf(US[k], Ws_w[tid * KRDU_ + k], z);
    z += Ws_b[tid];
    zs[tid] = z;

    auto blockReduce = [&](float v) -> float {
        red[tid] = v; __syncthreads();
        for (int s = 128; s > 0; s >>= 1) {
            if (tid < s) red[tid] += red[tid + s];
            __syncthreads();
        }
        float r = red[0]; __syncthreads();
        return r;
    };

    const float s_dot  = blockReduce(z * det_w[tid]);
    const float znorm2 = blockReduce(z * z);
    const float aent   = blockReduce(tid < V_ ? alpha[tid] * logf(alpha[tid] + EPS_) : 0.f);
    const float pent   = blockReduce(tid < 8  ? pi[tid]    * logf(pi[tid]    + EPS_) : 0.f);

    const float s       = s_dot + det_b[0];
    const float C_alpha = 1.f - (-aent) / logf((float)V_);
    const float u_pi    = (-pent) / logf(8.f);
    const float tau     = tau_c0[0] - lamU[0]*nu[0] - lamC[0]*C_alpha
                        + lamP[0]*risk[0] + lamPi[0]*u_pi;
    const float p_det   = 1.f / (1.f + expf(-(s - tau)));

    if (tid == 0) { out[O_PDET] = p_det; out[O_TAU] = tau; ws[DS_] = p_det; }

    // normalized zs -> workspace (used by the GEMM epilogue)
    ws[tid] = z / fmaxf(sqrtf(znorm2), 1e-12f);

    if (tid < KA_) {
        float y = cls_b[tid];
        for (int j = 0; j < DS_; ++j) y = fmaf(zs[j], cls_w[tid * DS_ + j], y);
        out[O_YHAT + tid] = 1.f / (1.f + expf(-y));
    }
    if (tid < V_) out[O_RVIEW + tid] = -0.5f * alpha[tid];
}

// ---------------------------------------------------------------------------
// Kernel 2: Z = H @ Wwin^T via v_wmma_f32_16x16x32_bf16, fused into
// e[row] = (Z_row . zs_n) / max(||Z_row||, 1e-12). Z never hits memory.
//
// Per wave: the full 16x256 A strip is held in registers (8 fragments,
// 64 VGPRs). N tiles are processed two at a time with transient v8f
// accumulators, folded straight into per-row num/nrm partials. B (Wwin^T,
// bf16) lives in 128 KB LDS in WMMA B-fragment order; the fill is fully
// vectorized (b128 stores). Strips are assigned contiguously per wave so
// each wave streams a sequential 128 KB of H, with a prefetch one strip
// ahead.
// ---------------------------------------------------------------------------
__global__ __launch_bounds__(256, 1) void gemm_e_kernel(
    const float* __restrict__ H, const float* __restrict__ Wwin,
    const float* __restrict__ zsn, float* __restrict__ e_out)
{
    extern __shared__ __align__(32) char smem_raw[];
    __bf16* ldsB = reinterpret_cast<__bf16*>(smem_raw); // 8*16*32*16 = 65536 elems

    const int tid = threadIdx.x;

    // ---- fill B fragments, vectorized: chunk c = frag*32 + lane holds the 16
    // bf16 elements (32 contiguous bytes) of that fragment's lane, which map
    // to 16 consecutive k of Wwin[n][k]:
    //   frag = kt*16+nt ; n = nt*16 + (lane&15) ; k = kt*32 + (lane>>4)*16 + i
#pragma unroll
    for (int it = 0; it < 16; ++it) {
        const int c    = it * 256 + tid;      // chunk id, 0..4095
        const int lane = c & 31;
        const int frag = c >> 5;              // kt*16 + nt
        const int kt   = frag >> 4;
        const int nt   = frag & 15;
        const int n    = nt * 16 + (lane & 15);
        const int kb   = kt * 32 + (lane >> 4) * 16;
        const float4* wp = reinterpret_cast<const float4*>(Wwin + n * DA_ + kb);
        const float4 w0 = wp[0], w1 = wp[1], w2 = wp[2], w3 = wp[3];
        v16bf b;
        b[0]  = (__bf16)w0.x; b[1]  = (__bf16)w0.y; b[2]  = (__bf16)w0.z; b[3]  = (__bf16)w0.w;
        b[4]  = (__bf16)w1.x; b[5]  = (__bf16)w1.y; b[6]  = (__bf16)w1.z; b[7]  = (__bf16)w1.w;
        b[8]  = (__bf16)w2.x; b[9]  = (__bf16)w2.y; b[10] = (__bf16)w2.z; b[11] = (__bf16)w2.w;
        b[12] = (__bf16)w3.x; b[13] = (__bf16)w3.y; b[14] = (__bf16)w3.z; b[15] = (__bf16)w3.w;
        *reinterpret_cast<v16bf*>(ldsB + (c << 4)) = b;
    }
    __syncthreads();

    const int wave = tid >> 5;
    const int lane = tid & 31;
    const int half = lane >> 4;
    const int lrow = lane & 15;
    const int gw   = blockIdx.x * 8 + wave;   // global wave id, 0..4095

    // per-lane copy of zs_n (column n = nt*16 + lrow)
    float zf[16];
#pragma unroll
    for (int nt = 0; nt < 16; ++nt) zf[nt] = zsn[nt * 16 + lrow];

    // 32768 strips total; each wave owns 8 consecutive strips = 128 KB of H
    for (int sidx = 0; sidx < 8; ++sidx) {
        // fence: stop LICM from hoisting LDS fragment loads across strips
        asm volatile("" ::: "memory");

        const int strip = gw * 8 + sidx;
        const float* hrow = H + ((size_t)strip * 16 + lrow) * DA_;

        // prefetch this lane's row of the next strip (gfx1250 global_prefetch)
        if (sidx < 7) __builtin_prefetch(hrow + 16 * DA_, 0, 1);

        // ---- load + convert the whole 16x256 A strip (8 fragments, 64 VGPRs)
        // A order used here (contiguous k, same as B fill): lanes 0-15 hold K = kt*32 + {0..15},
        // lanes 16-31 hold K = kt*32 + {16..31}.
        v16bf afrag[8];
#pragma unroll
        for (int kt = 0; kt < 8; ++kt) {
            const int kb = kt * 32 + half * 16;
            const float4* p0 = reinterpret_cast<const float4*>(hrow + kb);
            const float4* p1 = reinterpret_cast<const float4*>(hrow + kb + 8);
            const float4 f0 = p0[0], f1 = p0[1], f2 = p1[0], f3 = p1[1];
            v16bf a;
            a[0]  = (__bf16)f0.x; a[1]  = (__bf16)f0.y; a[2]  = (__bf16)f0.z; a[3]  = (__bf16)f0.w;
            a[4]  = (__bf16)f1.x; a[5]  = (__bf16)f1.y; a[6]  = (__bf16)f1.z; a[7]  = (__bf16)f1.w;
            a[8]  = (__bf16)f2.x; a[9]  = (__bf16)f2.y; a[10] = (__bf16)f2.z; a[11] = (__bf16)f2.w;
            a[12] = (__bf16)f3.x; a[13] = (__bf16)f3.y; a[14] = (__bf16)f3.z; a[15] = (__bf16)f3.w;
            afrag[kt] = a;
        }

        float num[8], nrm[8];
#pragma unroll
        for (int j = 0; j < 8; ++j) { num[j] = 0.f; nrm[j] = 0.f; }

        // ---- N tiles two at a time; accumulators are transient
#pragma unroll
        for (int np = 0; np < 8; ++np) {
            const int nt0 = np * 2;
            const int nt1 = np * 2 + 1;
            v8f acc0, acc1;
#pragma unroll
            for (int j = 0; j < 8; ++j) { acc0[j] = 0.f; acc1[j] = 0.f; }

#pragma unroll
            for (int kt = 0; kt < 8; ++kt) {
                const v16bf b0 = *reinterpret_cast<const v16bf*>(
                    ldsB + ((((kt * 16 + nt0) * 32) + lane) << 4));
                const v16bf b1 = *reinterpret_cast<const v16bf*>(
                    ldsB + ((((kt * 16 + nt1) * 32) + lane) << 4));
                acc0 = __builtin_amdgcn_wmma_f32_16x16x32_bf16(
                    false, afrag[kt], false, b0, (short)0, acc0, false, false);
                acc1 = __builtin_amdgcn_wmma_f32_16x16x32_bf16(
                    false, afrag[kt], false, b1, (short)0, acc1, false, false);
            }

            const float z0 = zf[nt0], z1 = zf[nt1];
#pragma unroll
            for (int j = 0; j < 8; ++j) {
                num[j] = fmaf(acc0[j], z0, num[j]);
                nrm[j] = fmaf(acc0[j], acc0[j], nrm[j]);
                num[j] = fmaf(acc1[j], z1, num[j]);
                nrm[j] = fmaf(acc1[j], acc1[j], nrm[j]);
            }
        }

        // epilogue: C/D layout -> VGPR j, lanes 0-15: M=j ; lanes 16-31: M=j+8
#pragma unroll
        for (int j = 0; j < 8; ++j) {
            float n0 = num[j], n1 = nrm[j];
#pragma unroll
            for (int mk = 1; mk < 16; mk <<= 1) {
                n0 += __shfl_xor(n0, mk, 32);
                n1 += __shfl_xor(n1, mk, 32);
            }
            if (lrow == 0) {
                const size_t row = (size_t)strip * 16 + half * 8 + j;
                e_out[row] = n0 / fmaxf(sqrtf(n1), 1e-12f);
            }
        }
    }
}

// ---------------------------------------------------------------------------
// Kernel 3: E_view[v] = mean over T of e[v,:]
// ---------------------------------------------------------------------------
__global__ __launch_bounds__(256) void eview_kernel(
    const float* __restrict__ e, float* __restrict__ out)
{
    __shared__ float red[256];
    const int v = blockIdx.x, tid = threadIdx.x;
    float s = 0.f;
    for (int t = tid; t < T_; t += 256) s += e[(size_t)v * T_ + t];
    red[tid] = s; __syncthreads();
    for (int st = 128; st > 0; st >>= 1) {
        if (tid < st) red[tid] += red[tid + st];
        __syncthreads();
    }
    if (tid == 0) out[O_EVIEW + v] = red[0] / (float)T_;
}

// ---------------------------------------------------------------------------
// Kernel 4: run-length starts/ends from e and r_view
// ---------------------------------------------------------------------------
__global__ __launch_bounds__(256) void interval_kernel(
    const float* __restrict__ e, const float* __restrict__ alpha,
    float* __restrict__ starts, float* __restrict__ ends)
{
    const int idx = blockIdx.x * blockDim.x + threadIdx.x;
    if (idx >= V_ * T_) return;
    const int v = idx / T_;
    const int t = idx - v * T_;
    const float r = -0.5f * alpha[v];
    const bool I    = e[idx] >= r;
    const bool prev = (t > 0)      ? (e[idx - 1] >= r) : false;
    const bool nxt  = (t < T_ - 1) ? (e[idx + 1] >= r) : false;
    starts[idx] = (I && !prev) ? 1.f : 0.f;
    ends[idx]   = (I && !nxt)  ? 1.f : 0.f;
}

// ---------------------------------------------------------------------------
// Kernel 5: I_view + flag_unknown
// ---------------------------------------------------------------------------
__global__ void final_kernel(float* __restrict__ out, const float* __restrict__ ws)
{
    __shared__ int anyI;
    const int v = threadIdx.x;
    if (v == 0) anyI = 0;
    __syncthreads();
    const bool Iv = out[O_EVIEW + v] >= out[O_RVIEW + v];
    out[O_IVIEW + v] = Iv ? 1.f : 0.f;
    if (Iv) atomicOr(&anyI, 1);
    __syncthreads();
    if (v == 0) {
        const float p = ws[DS_];
        out[O_FLAG] = ((p >= 0.5f) && (anyI == 0)) ? 1.f : 0.f;
    }
}

// ---------------------------------------------------------------------------
extern "C" void kernel_launch(void* const* d_in, const int* in_sizes, int n_in,
                              void* d_out, int out_size, void* d_ws, size_t ws_size,
                              hipStream_t stream)
{
    const float* h_aligned = (const float*)d_in[0];
    const float* alpha     = (const float*)d_in[1];
    const float* pi        = (const float*)d_in[2];
    const float* US        = (const float*)d_in[3];
    // d_in[4], d_in[5] (Wc_w, Wc_b) feed zc which is dead in the reference
    const float* Ws_w      = (const float*)d_in[6];
    const float* Ws_b      = (const float*)d_in[7];
    const float* det_w     = (const float*)d_in[8];
    const float* det_b     = (const float*)d_in[9];
    const float* cls_w     = (const float*)d_in[10];
    const float* cls_b     = (const float*)d_in[11];
    const float* Wwin      = (const float*)d_in[12];
    const float* tau_c0    = (const float*)d_in[13];
    const float* lamU      = (const float*)d_in[14];
    const float* lamC      = (const float*)d_in[15];
    const float* lamP      = (const float*)d_in[16];
    const float* lamPi     = (const float*)d_in[17];
    const float* nu        = (const float*)d_in[18];
    const float* risk      = (const float*)d_in[19];

    float* out = (float*)d_out;
    float* ws  = (float*)d_ws;

    head_kernel<<<1, 256, 0, stream>>>(alpha, pi, US, Ws_w, Ws_b, det_w, det_b,
                                       cls_w, cls_b, tau_c0, lamU, lamC, lamP,
                                       lamPi, nu, risk, out, ws);

    // 128 KB dynamic LDS for the pre-swizzled bf16 B fragments
    const size_t ldsBytes = (size_t)8 * 16 * 32 * 16 * 2;
    gemm_e_kernel<<<512, 256, ldsBytes, stream>>>(h_aligned, Wwin, ws, out + O_E);

    eview_kernel<<<V_, 256, 0, stream>>>(out + O_E, out);
    interval_kernel<<<(V_ * T_ + 255) / 256, 256, 0, stream>>>(
        out + O_E, alpha, out + O_STARTS, out + O_ENDS);
    final_kernel<<<1, V_, 0, stream>>>(out, ws);
}
